// STB_63548336112007
// MI455X (gfx1250) — matrix operand order repaired and checked by reading, not compiled, not run on hardware
//
#include <hip/hip_runtime.h>
#include <stdint.h>

#define TB 16
#define HH 96
#define WDIM 96
#define CDIM 256
#define SHF 3
#define NHD 8
#define HD 32
#define NTOK 36            // 6*6
#define NWIN 256           // (96/6)^2 per batch
#define MROWS (TB * NWIN * NTOK)   // 147456
#define SCALE_Q 5.656854249492381f     // sqrt(32), faithful to source
#define ALPHA_DN 1.8612097182041991f   // (2*6)^0.25

typedef __attribute__((ext_vector_type(16))) __bf16 v16bf;
typedef __attribute__((ext_vector_type(8)))  float  v8f;
typedef __attribute__((ext_vector_type(4)))  uint32_t v4u;
typedef __attribute__((ext_vector_type(8)))  int      v8i;
typedef __attribute__((ext_vector_type(4)))  int      v4i;

union BF8pack { __bf16 h[8]; uint4 u4; };

__device__ inline v8f v8f_zero() {
  v8f v;
#pragma unroll
  for (int i = 0; i < 8; ++i) v[i] = 0.f;
  return v;
}

__device__ inline v8f wmma_bf16(v16bf a, v16bf b, v8f c) {
  return __builtin_amdgcn_wmma_f32_16x16x32_bf16(false, a, false, b, (short)0, c,
                                                 false, false);
}

// Async global->LDS 16B copy, tracked by ASYNCcnt (CDNA5).
__device__ inline void async_copy_b128(uint32_t lds, const __bf16* g) {
  asm volatile("global_load_async_to_lds_b128 %0, %1, off"
               :
               : "v"(lds), "v"((uint64_t)(uintptr_t)g)
               : "memory");
}
__device__ inline void wait_asynccnt0() {
  asm volatile("s_wait_asynccnt 0x0" ::: "memory");
}

// TDM: load a 2D tile (rowlen bf16 elements x nrows, contiguous rows) into LDS,
// inserting 16B of LDS padding after every 64B (rowlen=32 -> LDS stride 40 halves).
// D# group0/group1 bitfields per CDNA5 ISA 8.3/8.4. 6-arg builtin (clang-23 form).
__device__ inline void tdm_load_tile_pad(const __bf16* gsrc, uint32_t lds_addr,
                                         int rowlen, int nrows) {
  uint64_t ga = (uint64_t)(uintptr_t)gsrc;
  v4u g0;
  g0[0] = 1u;                                   // count=1, user mode
  g0[1] = lds_addr;                             // lds_addr [63:32]
  g0[2] = (uint32_t)ga;                         // global_addr [95:64]
  g0[3] = (uint32_t)((ga >> 32) & 0x01FFFFFFu)  // global_addr [120:96]
          | (2u << 30);                         // type=2 (image)
  v8i g1;
  g1[0] = (1 << 16)        // data_size = 2B
        | (1 << 20)        // pad_enable
        | (3 << 22)        // pad_interval: 16 DWORDs = 64B
        | (3 << 25);       // pad_amount: 4 DWORDs = 16B
  g1[1] = (rowlen & 0xFFFF) << 16;              // tensor_dim0[15:0] at bits 63:48
  g1[2] = (nrows & 0xFFFF) << 16;               // tensor_dim1[15:0] (bits 95:80)
  g1[3] = (rowlen & 0xFFFF) << 16;              // tile_dim0 at bits 127:112
  g1[4] = (nrows & 0xFFFF);                     // tile_dim1; tile_dim2 = 0
  g1[5] = rowlen;                               // tensor_dim0_stride low32
  g1[6] = ((rowlen * nrows) & 0xFFFF) << 16;    // dim0_stride hi=0 | dim1_stride lo16
  g1[7] = 0;                                    // dim1_stride hi
  v4i z4;
  z4[0] = 0; z4[1] = 0; z4[2] = 0; z4[3] = 0;
  v8i z8;
#pragma unroll
  for (int i = 0; i < 8; ++i) z8[i] = 0;
  __builtin_amdgcn_tensor_load_to_lds(g0, g1, z4, z4, z8, 0);
}

// A fragment (16x32 bf16): lane 0-15 -> M=lane, K=0..15
// lane 16-31 -> M=lane-16, K=16..31. Source row-major, K contiguous.
__device__ inline v16bf load_a_frag(const __bf16* base, int stride, int m0, int k0, int lane) {
  const int half = lane >> 4, mr = lane & 15;
  const __bf16* row = base + (m0 + mr) * stride + k0 + 16 * half;
  union { v16bf v; uint32_t u[8]; } r;
#pragma unroll
  for (int j = 0; j < 4; ++j) {
    r.u[j]     = *(const uint32_t*)(row + 2 * j);
    r.u[j + 4] = *(const uint32_t*)(row + 8 + 2 * j);
  }
  return r.v;
}

// B fragment (32x16 bf16) from B^T layout [N][K]: lane 0-15 -> N=lane, K=0..15;
// lane 16-31 -> N=lane-16, K=16..31. K contiguous pairs per dword.
__device__ inline v16bf load_b_frag(const __bf16* base, int stride, int n0, int k0, int lane) {
  const int half = lane >> 4, nc = lane & 15;
  const __bf16* row = base + (n0 + nc) * stride + k0 + 16 * half;
  union { v16bf v; uint32_t u[8]; } r;
#pragma unroll
  for (int j = 0; j < 8; ++j) r.u[j] = *(const uint32_t*)(row + 2 * j);
  return r.v;
}

// ---------------- prep kernels ----------------

// f32 [K][N] -> bf16 [N][K]
__global__ void wconv_kernel(const float* __restrict__ in, __bf16* __restrict__ out,
                             int K, int N) {
  int idx = blockIdx.x * 256 + threadIdx.x;
  if (idx >= K * N) return;
  int n = idx / K, k = idx - n * K;
  out[idx] = (__bf16)in[(size_t)k * N + n];
}

// biasf[h][i][j] = table[((iy-jy+5)*11 + (ix-jx+5))][h]
__global__ void bias_kernel(const float* __restrict__ tbl, float* __restrict__ bf) {
  int idx = blockIdx.x * 256 + threadIdx.x;
  if (idx >= NHD * NTOK * NTOK) return;
  int h = idx / (NTOK * NTOK);
  int rem = idx % (NTOK * NTOK);
  int i = rem / NTOK, j = rem % NTOK;
  int ih = i / 6, iw = i % 6, jh = j / 6, jw = j % 6;
  int rel = (ih - jh + 5) * 11 + (iw - jw + 5);
  bf[idx] = tbl[rel * NHD + h];
}

// x (B, 9216, 256) f32 -> roll(-3,-3) + window partition -> xw bf16 [MROWS][256]
__global__ __launch_bounds__(256) void xgather_kernel(const float* __restrict__ x,
                                                      __bf16* __restrict__ xw) {
  int idx = blockIdx.x * 256 + threadIdx.x;  // [0, MROWS*32)
  int row = idx >> 5;
  int cb = (idx & 31) * 8;
  int win = row / NTOK, n = row % NTOK;
  int b = win / NWIN, w = win % NWIN;
  int wy = w / 16, wx = w % 16;
  int ty = n / 6, tx = n % 6;
  int y = (wy * 6 + ty + SHF) % HH;
  int xc = (wx * 6 + tx + SHF) % WDIM;
  const float* src = x + ((size_t)b * HH * WDIM + y * WDIM + xc) * CDIM + cb;
  float4 f0 = ((const float4*)src)[0];
  float4 f1 = ((const float4*)src)[1];
  BF8pack pk;
  pk.h[0] = (__bf16)f0.x; pk.h[1] = (__bf16)f0.y; pk.h[2] = (__bf16)f0.z; pk.h[3] = (__bf16)f0.w;
  pk.h[4] = (__bf16)f1.x; pk.h[5] = (__bf16)f1.y; pk.h[6] = (__bf16)f1.z; pk.h[7] = (__bf16)f1.w;
  *(uint4*)(xw + (size_t)row * CDIM + cb) = pk.u4;
}

// ---------------- GEMM with fused epilogues ----------------
// C[m][n] = sum_k A[m][k] * Bt[n][k]; 256 threads, 128x128 tile, BK=32.
// Double-buffered LDS fed by async global->LDS DMA (ASYNCcnt).
// EPI: 0=QKV scatter, 1=proj scatter+shortcut, 2=FC1+GELU(bf16), 3=FC2+resid(f32)
template <int EPI>
__global__ __launch_bounds__(256) void gemm_bf16_kernel(
    const __bf16* __restrict__ A, const __bf16* __restrict__ Bt,
    int M, int N, int K, const float* __restrict__ cbias,
    __bf16* __restrict__ ob0, __bf16* __restrict__ ob1, __bf16* __restrict__ ob2,
    float* __restrict__ of0, const float* __restrict__ af0,
    const __bf16* __restrict__ ab0) {
  __shared__ __bf16 As[2][128 * 40];
  __shared__ __bf16 Bs[2][128 * 40];
  const int tid = threadIdx.x;
  const int lane = tid & 31, wid = tid >> 5;
  const int wm = wid >> 1, wn = wid & 1;   // 4 x 2 waves -> 32x64 per wave
  const int mblk = blockIdx.y * 128, nblk = blockIdx.x * 128;
  const int arow = tid >> 1, apart = (tid & 1) * 16;

  const __bf16* agp = A + (size_t)(mblk + arow) * K + apart;
  const __bf16* bgp = Bt + (size_t)(nblk + arow) * K + apart;
  uint32_t al[2], bl[2];
#pragma unroll
  for (int i = 0; i < 2; ++i) {
    al[i] = (uint32_t)(uintptr_t)&As[i][arow * 40 + apart];
    bl[i] = (uint32_t)(uintptr_t)&Bs[i][arow * 40 + apart];
  }

  v8f acc[2][4];
#pragma unroll
  for (int i = 0; i < 2; ++i)
#pragma unroll
    for (int j = 0; j < 4; ++j) acc[i][j] = v8f_zero();

  // prologue: async-fill buffer 0
  async_copy_b128(al[0], agp);
  async_copy_b128(al[0] + 16, agp + 8);
  async_copy_b128(bl[0], bgp);
  async_copy_b128(bl[0] + 16, bgp + 8);
  wait_asynccnt0();
  __syncthreads();

  int buf = 0;
  for (int kk = 0; kk < K; kk += 32, buf ^= 1) {
    if (kk + 32 < K) {  // prefetch next chunk into other buffer
      const __bf16* an = agp + kk + 32;
      const __bf16* bn = bgp + kk + 32;
      async_copy_b128(al[buf ^ 1], an);
      async_copy_b128(al[buf ^ 1] + 16, an + 8);
      async_copy_b128(bl[buf ^ 1], bn);
      async_copy_b128(bl[buf ^ 1] + 16, bn + 8);
    }

    v16bf af[2], bfr[4];
#pragma unroll
    for (int mt = 0; mt < 2; ++mt)
      af[mt] = load_a_frag(As[buf], 40, wm * 32 + mt * 16, 0, lane);
#pragma unroll
    for (int nt = 0; nt < 4; ++nt)
      bfr[nt] = load_b_frag(Bs[buf], 40, wn * 64 + nt * 16, 0, lane);
#pragma unroll
    for (int mt = 0; mt < 2; ++mt)
#pragma unroll
      for (int nt = 0; nt < 4; ++nt) acc[mt][nt] = wmma_bf16(af[mt], bfr[nt], acc[mt][nt]);

    wait_asynccnt0();
    __syncthreads();
  }

  const int half = lane >> 4, nc = lane & 15;
#pragma unroll
  for (int mt = 0; mt < 2; ++mt) {
#pragma unroll
    for (int nt = 0; nt < 4; ++nt) {
      const int gnn = nblk + wn * 64 + nt * 16 + nc;
#pragma unroll
      for (int r = 0; r < 8; ++r) {
        const int gmm = mblk + wm * 32 + mt * 16 + 8 * half + r;
        float val = acc[mt][nt][r] + cbias[gnn];
        if (EPI == 0) {  // QKV: scatter q(scaled)/k row-major, v transposed
          int s = gnn >> 8, hn = (gnn >> 5) & 7, dd = gnn & 31;
          int win = gmm / NTOK, tok = gmm % NTOK;
          if (s == 0)
            ob0[((size_t)(win * NHD + hn) * NTOK + tok) * HD + dd] = (__bf16)(val * SCALE_Q);
          else if (s == 1)
            ob1[((size_t)(win * NHD + hn) * NTOK + tok) * HD + dd] = (__bf16)val;
          else
            ob2[((size_t)(win * NHD + hn) * HD + dd) * NTOK + tok] = (__bf16)val;
        } else if (EPI == 1) {  // proj: buggy window_reverse (plain reshape) + roll + shortcut
          int b = gmm / (NWIN * NTOK);
          int qq = gmm % (NWIN * NTOK);
          int a1 = qq / 576, rem = qq % 576;
          int a2 = rem / 96, rem2 = rem % 96;
          int a3 = rem2 / 6, a4 = rem2 % 6;
          int y = (a1 * 6 + a2 + SHF) % HH;
          int xx = (a3 * 6 + a4 + SHF) % WDIM;
          size_t di = ((size_t)b * HH * WDIM + y * WDIM + xx) * CDIM + gnn;
          of0[di] = af0[di] * ALPHA_DN + val;
        } else if (EPI == 2) {  // FC1 + exact GELU -> bf16
          float g = 0.5f * val * (1.0f + erff(val * 0.70710678118654752f));
          ob0[(size_t)gmm * N + gnn] = (__bf16)g;
        } else {  // FC2 + ALPHA * x1 residual -> f32
          of0[(size_t)gmm * CDIM + gnn] =
              val + ALPHA_DN * (float)ab0[(size_t)gmm * CDIM + gnn];
        }
      }
    }
  }
}

// ---------------- attention: one wave per (window, head) ----------------
__global__ __launch_bounds__(32) void attn_kernel(
    const __bf16* __restrict__ q, const __bf16* __restrict__ k,
    const __bf16* __restrict__ vT, const float* __restrict__ biasf,
    __bf16* __restrict__ out) {
  const int wh = blockIdx.x;  // win*8 + h
  const int win = wh >> 3, h = wh & 7;
  const int lane = threadIdx.x;
  __shared__ __bf16 qs[48 * 40];
  __shared__ __bf16 ks[48 * 40];
  __shared__ float Sf[48 * 52];
  __shared__ __bf16 Ps[48 * 72];
  __shared__ __bf16 vs[32 * 72];
  __shared__ int gsh[36];

  for (int i = lane; i < 48 * 40; i += 32) { qs[i] = (__bf16)0.f; ks[i] = (__bf16)0.f; }
  for (int i = lane; i < 48 * 72; i += 32) Ps[i] = (__bf16)0.f;
  for (int i = lane; i < 32 * 72; i += 32) vs[i] = (__bf16)0.f;
  __syncthreads();

  const size_t qb = (size_t)wh * NTOK * HD;
  // q/k tiles via Tensor Data Mover: 36 rows x 32 bf16, LDS-padded 16B per 64B
  // so the landed layout is exactly stride-40-halves (bank-conflict-free frags).
  tdm_load_tile_pad(q + qb, (uint32_t)(uintptr_t)&qs[0], HD, NTOK);
  tdm_load_tile_pad(k + qb, (uint32_t)(uintptr_t)&ks[0], HD, NTOK);

  for (int t = lane; t < HD * NTOK; t += 32) {
    int d = t / NTOK, n = t - d * NTOK;
    vs[d * 72 + n] = vT[qb + t];
  }
  for (int t = lane; t < NTOK; t += 32) {  // shift-mask group ids
    int ty = t / 6, tx = t % 6;
    int wloc = win % NWIN;
    int y = (wloc / 16) * 6 + ty, xx = (wloc % 16) * 6 + tx;
    int ry = y < 90 ? 0 : (y < 93 ? 1 : 2);
    int rx = xx < 90 ? 0 : (xx < 93 ? 1 : 2);
    gsh[t] = ry * 3 + rx;
  }
  __builtin_amdgcn_s_wait_tensorcnt(0);
  __syncthreads();

  // S = q @ k^T  (K = 32 exact -> one WMMA per 16x16 tile)
  const int half = lane >> 4, nc = lane & 15;
  {
    v16bf af[3], bfr[3];
#pragma unroll
    for (int mt = 0; mt < 3; ++mt) af[mt] = load_a_frag(qs, 40, mt * 16, 0, lane);
#pragma unroll
    for (int nt = 0; nt < 3; ++nt) bfr[nt] = load_b_frag(ks, 40, nt * 16, 0, lane);
#pragma unroll
    for (int mt = 0; mt < 3; ++mt) {
#pragma unroll
      for (int nt = 0; nt < 3; ++nt) {
        v8f s = wmma_bf16(af[mt], bfr[nt], v8f_zero());
#pragma unroll
        for (int r = 0; r < 8; ++r)
          Sf[(mt * 16 + 8 * half + r) * 52 + nt * 16 + nc] = s[r];
      }
    }
  }
  __syncthreads();

  // bias + mask + softmax over valid 36 keys; emit P as bf16 (K-padded to 64 with zeros)
  const float* brow = biasf + h * NTOK * NTOK;
  for (int m = lane; m < NTOK; m += 32) {
    int gi = gsh[m];
    float mx = -1e30f;
    float* srow = Sf + m * 52;
    const float* bi = brow + m * NTOK;
    for (int j = 0; j < NTOK; ++j) {
      float sv = srow[j] + bi[j] + ((gi == gsh[j]) ? 0.f : -100.f);
      srow[j] = sv;
      mx = fmaxf(mx, sv);
    }
    float sum = 0.f;
    for (int j = 0; j < NTOK; ++j) {
      float e = __expf(srow[j] - mx);
      srow[j] = e;
      sum += e;
    }
    float inv = 1.f / sum;
    for (int j = 0; j < NTOK; ++j) Ps[m * 72 + j] = (__bf16)(srow[j] * inv);
  }
  __syncthreads();

  // O = P @ v  (K padded 36->64, N = 32)
  v8f acc[3][2];
#pragma unroll
  for (int mt = 0; mt < 3; ++mt)
#pragma unroll
    for (int nt = 0; nt < 2; ++nt) acc[mt][nt] = v8f_zero();
#pragma unroll
  for (int kk = 0; kk < 2; ++kk) {
    v16bf a2[3], b2[2];
#pragma unroll
    for (int mt = 0; mt < 3; ++mt) a2[mt] = load_a_frag(Ps, 72, mt * 16, kk * 32, lane);
#pragma unroll
    for (int nt = 0; nt < 2; ++nt) b2[nt] = load_b_frag(vs, 72, nt * 16, kk * 32, lane);
#pragma unroll
    for (int mt = 0; mt < 3; ++mt)
#pragma unroll
      for (int nt = 0; nt < 2; ++nt) acc[mt][nt] = wmma_bf16(a2[mt], b2[nt], acc[mt][nt]);
  }
#pragma unroll
  for (int mt = 0; mt < 3; ++mt) {
#pragma unroll
    for (int nt = 0; nt < 2; ++nt) {
#pragma unroll
      for (int r = 0; r < 8; ++r) {
        int m = mt * 16 + 8 * half + r;
        if (m < NTOK)
          out[((size_t)win * NTOK + m) * CDIM + h * HD + nt * 16 + nc] =
              (__bf16)acc[mt][nt][r];
      }
    }
  }
}

// ---------------- LayerNorm over C=256, wave per row ----------------
template <bool BF16OUT>
__global__ __launch_bounds__(256) void ln_kernel(const float* __restrict__ in,
                                                 const float* __restrict__ g,
                                                 const float* __restrict__ bta,
                                                 float* __restrict__ of,
                                                 __bf16* __restrict__ ob) {
  int row = blockIdx.x * 8 + (threadIdx.x >> 5);
  int lane = threadIdx.x & 31;
  const float* p = in + (size_t)row * CDIM + lane * 8;
  float4 v0 = ((const float4*)p)[0];
  float4 v1 = ((const float4*)p)[1];
  float vals[8] = {v0.x, v0.y, v0.z, v0.w, v1.x, v1.y, v1.z, v1.w};
  float s = 0.f, sq = 0.f;
#pragma unroll
  for (int j = 0; j < 8; ++j) { s += vals[j]; sq += vals[j] * vals[j]; }
#pragma unroll
  for (int off = 16; off > 0; off >>= 1) {
    s += __shfl_xor(s, off, 32);
    sq += __shfl_xor(sq, off, 32);
  }
  float mean = s * (1.f / CDIM);
  float var = sq * (1.f / CDIM) - mean * mean;
  float rstd = rsqrtf(var + 1e-5f);
  int cb = lane * 8;
#pragma unroll
  for (int j = 0; j < 8; ++j) {
    float o = (vals[j] - mean) * rstd * g[cb + j] + bta[cb + j];
    if (BF16OUT)
      ob[(size_t)row * CDIM + cb + j] = (__bf16)o;
    else
      of[(size_t)row * CDIM + cb + j] = o;
  }
}

// ---------------- host launch ----------------
extern "C" void kernel_launch(void* const* d_in, const int* in_sizes, int n_in,
                              void* d_out, int out_size, void* d_ws, size_t ws_size,
                              hipStream_t stream) {
  const float* x      = (const float*)d_in[0];
  const float* qkv_w  = (const float*)d_in[1];
  const float* qkv_b  = (const float*)d_in[2];
  const float* btab   = (const float*)d_in[3];
  const float* proj_w = (const float*)d_in[4];
  const float* proj_b = (const float*)d_in[5];
  const float* n1w    = (const float*)d_in[6];
  const float* n1b    = (const float*)d_in[7];
  const float* n2w    = (const float*)d_in[8];
  const float* n2b    = (const float*)d_in[9];
  const float* fc1_w  = (const float*)d_in[10];
  const float* fc1_b  = (const float*)d_in[11];
  const float* fc2_w  = (const float*)d_in[12];
  const float* fc2_b  = (const float*)d_in[13];

  char* w = (char*)d_ws;
  size_t o = 0;
  __bf16* qkv_wt = (__bf16*)(w + o); o += (size_t)196608 * 2;
  __bf16* proj_wt = (__bf16*)(w + o); o += (size_t)65536 * 2;
  __bf16* fc1_wt = (__bf16*)(w + o); o += (size_t)262144 * 2;
  __bf16* fc2_wt = (__bf16*)(w + o); o += (size_t)262144 * 2;
  float* biasf = (float*)(w + o); o += (size_t)10368 * 4;
  char* R = w + o; o += (size_t)MROWS * 1024 * 2;  // big region, aliased
  __bf16* xw  = (__bf16*)R;                               // also attn_out after QKV use
  __bf16* qd  = (__bf16*)(R + (size_t)MROWS * CDIM * 2);
  __bf16* kd  = (__bf16*)(R + (size_t)MROWS * CDIM * 2 * 2);
  __bf16* vTd = (__bf16*)(R + (size_t)MROWS * CDIM * 2 * 3);
  __bf16* hmid = (__bf16*)R;                              // FC1 hidden reuses whole region
  float* x1pre = (float*)(w + o); o += (size_t)MROWS * CDIM * 4;  // also x2pre
  __bf16* x1bf = (__bf16*)(w + o); o += (size_t)MROWS * CDIM * 2;

  // 1. weight convert/transpose (one-time, tiny)
  wconv_kernel<<<(196608 + 255) / 256, 256, 0, stream>>>(qkv_w, qkv_wt, 256, 768);
  wconv_kernel<<<(65536 + 255) / 256, 256, 0, stream>>>(proj_w, proj_wt, 256, 256);
  wconv_kernel<<<(262144 + 255) / 256, 256, 0, stream>>>(fc1_w, fc1_wt, 256, 1024);
  wconv_kernel<<<(262144 + 255) / 256, 256, 0, stream>>>(fc2_w, fc2_wt, 1024, 256);
  bias_kernel<<<(10368 + 255) / 256, 256, 0, stream>>>(btab, biasf);

  // 2. gather x -> shifted window layout, bf16
  xgather_kernel<<<MROWS * 32 / 256, 256, 0, stream>>>(x, xw);

  // 3. QKV GEMM (147456 x 768 x 256)
  gemm_bf16_kernel<0><<<dim3(768 / 128, MROWS / 128), 256, 0, stream>>>(
      xw, qkv_wt, MROWS, 768, 256, qkv_b, qd, kd, vTd, nullptr, nullptr, nullptr);

  // 4. windowed attention, one wave per (window, head)
  attn_kernel<<<TB * NWIN * NHD, 32, 0, stream>>>(qd, kd, vTd, biasf, xw);

  // 5. proj GEMM + window_reverse(buggy)+roll scatter + shortcut*ALPHA
  gemm_bf16_kernel<1><<<dim3(256 / 128, MROWS / 128), 256, 0, stream>>>(
      xw, proj_wt, MROWS, 256, 256, proj_b, nullptr, nullptr, nullptr, x1pre, x, nullptr);

  // 6. LN1 -> x1 bf16
  ln_kernel<true><<<MROWS / 8, 256, 0, stream>>>(x1pre, n1w, n1b, nullptr, x1bf);

  // 7. FC1 GEMM + GELU (147456 x 1024 x 256)
  gemm_bf16_kernel<2><<<dim3(1024 / 128, MROWS / 128), 256, 0, stream>>>(
      x1bf, fc1_wt, MROWS, 1024, 256, fc1_b, hmid, nullptr, nullptr, nullptr, nullptr,
      nullptr);

  // 8. FC2 GEMM + ALPHA*x1 residual (147456 x 256 x 1024) -> x2pre (aliases x1pre)
  gemm_bf16_kernel<3><<<dim3(256 / 128, MROWS / 128), 256, 0, stream>>>(
      hmid, fc2_wt, MROWS, 256, 1024, fc2_b, nullptr, nullptr, nullptr, x1pre, nullptr,
      x1bf);

  // 9. LN2 -> f32 output
  ln_kernel<false><<<MROWS / 8, 256, 0, stream>>>(x1pre, n2w, n2b, (float*)d_out, nullptr);
}
